// ViTOutput_8117488190193
// MI455X (gfx1250) — matrix operand order repaired and checked by reading, not compiled, not run on hardware
//
#include <hip/hip_runtime.h>
#include <stdint.h>

// ---------------------------------------------------------------------------
// ViTOutput fused kernel for gfx1250 (MI455X), BF16 WMMA path.
//   dense = (hs + move) @ W^T + b            [36928 x 768], K = 3072
//   out   = LayerScale(RPReLU(LN(dense) + AvgPool4(hs)))
// ---------------------------------------------------------------------------

typedef __attribute__((ext_vector_type(16))) __bf16 v16bf;
typedef __attribute__((ext_vector_type(8)))  float  v8f;
typedef __attribute__((ext_vector_type(2)))  float  v2f;
typedef __attribute__((ext_vector_type(2)))  __bf16 v2bf;

union V16 { v16bf v; uint4 q[2]; };

// Native f32 -> bf16 (backend selects v_cvt_pk_bf16_f32).
__device__ __forceinline__ uint32_t cvt_pk_bf16(float a, float b) {
  v2f   f = {a, b};
  v2bf  h = __builtin_convertvector(f, v2bf);
  union { v2bf h; uint32_t u; } c; c.h = h;
  return c.u;
}

// Pinned B-fragment load: two b128 into caller-visible regs (outputs only —
// tied asm operands are not supported by the gfx1250 backend).
#define LOAD_BFRAG(frag, base, byteoff)                                      \
  asm volatile("global_load_b128 %0, %2, off offset:%c3\n\t"                 \
               "global_load_b128 %1, %2, off offset:%c4"                     \
               : "=v"((frag).q[0]), "=v"((frag).q[1])                        \
               : "v"(base), "n"(byteoff), "n"((byteoff) + 16))

// Partial loadcnt wait, fenced with sched_barrier(0) so the scheduler cannot
// hoist the consuming WMMAs above the wait nor sink the loads below it.
#define WAIT_LOADCNT(n)                                                      \
  do {                                                                       \
    __builtin_amdgcn_sched_barrier(0);                                       \
    asm volatile("s_wait_loadcnt %c0" ::"n"(n));                             \
    __builtin_amdgcn_sched_barrier(0);                                       \
  } while (0)

// ---------------------------------------------------------------------------
// Pack W [768][3072] fp32 -> bf16 fragment-major layout:
//   Wp[((kt*48 + nt)*32 + lane)*16 + e]  with
//   n = nt*16 + (lane&15), k = kt*32 + (lane<16 ? 0 : 16) + e
// Contiguous k order (k = 16h + e), the same order as the A fragment: a wave loads its
// whole B fragment as one contiguous 32B chunk per lane (2x b128).
// ---------------------------------------------------------------------------
__global__ void pack_w_bf16(const float* __restrict__ W,
                            unsigned short* __restrict__ Wp) {
  const int t = blockIdx.x * blockDim.x + threadIdx.x;   // 0 .. 96*48*32-1
  const int lane = t & 31;
  const int nt   = (t >> 5) % 48;
  const int kt   = t / (32 * 48);
  if (kt >= 96) return;
  const int n     = nt * 16 + (lane & 15);
  const int kbase = kt * 32 + ((lane < 16) ? 0 : 16);
  const float* src = W + (size_t)n * 3072 + kbase;       // 64B contiguous read
  uint32_t* dst = (uint32_t*)(Wp + (size_t)t * 16);      // 32B contiguous write
#pragma unroll
  for (int e = 0; e < 8; ++e) dst[e] = cvt_pk_bf16(src[2 * e], src[2 * e + 1]);
}

// ---------------------------------------------------------------------------
// Main fused kernel. 256 threads = 8 waves. WG tile: 32 rows x 768 cols.
// Wave (msub, nq) = (wid>>2, wid&3): 16 rows x 192 cols = 12 WMMA tiles.
// ---------------------------------------------------------------------------
__global__ void __launch_bounds__(256, 1)
vit_out_fused(const float* __restrict__ hs,            // [36928][3072]
              const unsigned short* __restrict__ Wp,   // packed bf16 W
              const float* __restrict__ bias,          // [768]
              const float* __restrict__ move,          // [3072]
              const float* __restrict__ gamma,
              const float* __restrict__ beta,
              const float* __restrict__ move1,
              const float* __restrict__ pa,
              const float* __restrict__ move2,
              const float* __restrict__ lsa,
              const float* __restrict__ lsm,
              float* __restrict__ out)                 // [36928][768]
{
  constexpr int I    = 3072;
  constexpr int H    = 768;
  constexpr int MT   = 32;        // rows per workgroup
  constexpr int KT   = 32;        // K per iteration
  constexpr int NK   = I / KT;    // 96 K-steps
  constexpr int APAD = 40;        // padded A row in ushorts (80B: 16B-aligned,
                                  // 20-bank stride -> conflict-free b128 reads)

  extern __shared__ char smem[];
  float*          pooled = (float*)smem;                          // [32][768]
  unsigned short* aBuf   = (unsigned short*)(smem + MT * H * 4);  // [2][32][APAD]
  float*          rowSum = (float*)(smem + MT * H * 4 + 2 * MT * APAD * 2);
  float*          rowSq  = rowSum + MT;

  const int tid  = threadIdx.x;
  const int lane = tid & 31;
  const int wid  = tid >> 5;
  const int mrow = blockIdx.x * MT;

  if (tid < MT) { rowSum[tid] = 0.f; rowSq[tid] = 0.f; }

  // A-load mapping: thread -> (row 0..31, k-quad 0..7); one float4 per thread.
  const int lr = tid >> 3;
  const int kq = tid & 7;
  const float* hrow = hs + (size_t)(mrow + lr) * I + kq * 4;

  // Wave/WMMA mapping.
  const int msub = wid >> 2;                 // 0..1  (row half)
  const int ntb  = (wid & 3) * 12;           // first of 12 n-tiles
  const int rloc = msub * 16 + (lane & 15);  // A-fragment row for this lane
  const int c0   = (lane < 16) ? 0 : 16;     // A-fragment K sub-offset

  v8f acc[12];
  const v8f vzero = {0.f, 0.f, 0.f, 0.f, 0.f, 0.f, 0.f, 0.f};
#pragma unroll
  for (int t = 0; t < 12; ++t) acc[t] = vzero;

  // --- prologue: prefetch first A chunk into registers ---
  float4 x  = *(const float4*)(hrow);
  float4 mv = *(const float4*)(move + kq * 4);

  for (int kt = 0; kt < NK; ++kt) {
    const int buf = kt & 1;

    // ---- store phase: fp32 -> bf16 into LDS + AvgPool4 (raw hidden) ----
    pooled[lr * H + kt * 8 + kq] = (x.x + x.y + x.z + x.w) * 0.25f;
    {
      uint2 pk;
      pk.x = cvt_pk_bf16(x.x + mv.x, x.y + mv.y);
      pk.y = cvt_pk_bf16(x.z + mv.z, x.w + mv.w);
      *(uint2*)(aBuf + ((size_t)(buf * MT + lr)) * APAD + kq * 4) = pk;
    }

    // ---- prefetch next K-step's A chunk (latency hidden by WMMA phase) ----
    float4 xn, mvn;
    if (kt + 1 < NK) {
      xn  = *(const float4*)(hrow + (size_t)(kt + 1) * KT);
      mvn = *(const float4*)(move + (kt + 1) * KT + kq * 4);
    }

    __syncthreads();   // single barrier/iter; double buffer covers WAR hazard

    // ---- A fragment from LDS (contiguous k = c0 + s, same order as packed B) ----
    V16 a;
    const unsigned short* ap = aBuf + ((size_t)(buf * MT + rloc)) * APAD;
    a.q[0] = *(const uint4*)(ap + c0);        // K = c0 .. c0+7
    a.q[1] = *(const uint4*)(ap + c0 + 8);    // K = c0+8 .. c0+15

    // ---- B: issue ALL 24 b128 loads up-front (inline asm, pinned regs),
    //      then gate WMMA groups with partial s_wait_loadcnt 16 -> 8 -> 0.
    //      In-order loadcnt retirement makes these bounds safe no matter how
    //      the compiler interleaves its own loads. ----
    const uint4* bbase =
        (const uint4*)Wp + ((((size_t)kt * 48 + ntb) * 32 + lane) << 1);
    if (kt + 1 < NK)   // hint GL2: next K-step's B block
      __builtin_prefetch((const uint4*)Wp +
                         ((((size_t)(kt + 1) * 48 + ntb) * 32 + lane) << 1),
                         0, 1);
    V16 bk[12];
    LOAD_BFRAG(bk[0],  bbase, 0 * 128);
    LOAD_BFRAG(bk[1],  bbase, 1 * 128);
    LOAD_BFRAG(bk[2],  bbase, 2 * 128);
    LOAD_BFRAG(bk[3],  bbase, 3 * 128);
    LOAD_BFRAG(bk[4],  bbase, 4 * 128);
    LOAD_BFRAG(bk[5],  bbase, 5 * 128);
    LOAD_BFRAG(bk[6],  bbase, 6 * 128);
    LOAD_BFRAG(bk[7],  bbase, 7 * 128);
    LOAD_BFRAG(bk[8],  bbase, 8 * 128);
    LOAD_BFRAG(bk[9],  bbase, 9 * 128);
    LOAD_BFRAG(bk[10], bbase, 10 * 128);
    LOAD_BFRAG(bk[11], bbase, 11 * 128);

    WAIT_LOADCNT(16);                         // tiles 0..3 landed
#pragma unroll
    for (int g = 0; g < 4; ++g)
      acc[g] = __builtin_amdgcn_wmma_f32_16x16x32_bf16(
          false, a.v, false, bk[g].v, (short)0, acc[g], false, false);

    WAIT_LOADCNT(8);                          // tiles 4..7 landed
#pragma unroll
    for (int g = 4; g < 8; ++g)
      acc[g] = __builtin_amdgcn_wmma_f32_16x16x32_bf16(
          false, a.v, false, bk[g].v, (short)0, acc[g], false, false);

    WAIT_LOADCNT(0);                          // tiles 8..11 landed
#pragma unroll
    for (int g = 8; g < 12; ++g)
      acc[g] = __builtin_amdgcn_wmma_f32_16x16x32_bf16(
          false, a.v, false, bk[g].v, (short)0, acc[g], false, false);

    x  = xn;
    mv = mvn;
  }

  // ---- bias fold + per-row partial sums over this wave's 192 columns ----
  const int nwcol = (wid & 3) * 192;
  float bv[12];
#pragma unroll
  for (int t = 0; t < 12; ++t) bv[t] = bias[nwcol + t * 16 + (lane & 15)];

  float psum[8], psq[8];
#pragma unroll
  for (int j = 0; j < 8; ++j) { psum[j] = 0.f; psq[j] = 0.f; }
#pragma unroll
  for (int t = 0; t < 12; ++t) {
#pragma unroll
    for (int j = 0; j < 8; ++j) {
      const float v = acc[t][j] + bv[t];
      acc[t][j] = v;
      psum[j] += v;
      psq[j]  += v * v;
    }
  }
  // reduce across the 16 lanes of each half-wave (C-layout: cols = lanes)
#pragma unroll
  for (int j = 0; j < 8; ++j) {
#pragma unroll
    for (int m = 1; m <= 8; m <<= 1) {
      psum[j] += __shfl_xor(psum[j], m, 32);
      psq[j]  += __shfl_xor(psq[j],  m, 32);
    }
  }
  // combine the 4 waves sharing each row group via ds_add_f32
  if (lane == 0 || lane == 16) {
    const int rbase = msub * 16 + ((lane == 16) ? 8 : 0);
#pragma unroll
    for (int j = 0; j < 8; ++j) {
      atomicAdd(&rowSum[rbase + j], psum[j]);
      atomicAdd(&rowSq[rbase + j],  psq[j]);
    }
  }
  __syncthreads();

  // ---- fused LN + pooled add + RPReLU + LayerScale + store ----
  const float invH = 1.0f / 768.0f;
#pragma unroll
  for (int t = 0; t < 12; ++t) {
    const int col = nwcol + t * 16 + (lane & 15);
    const float g  = gamma[col], be = beta[col];
    const float m1 = move1[col], al = pa[col], m2 = move2[col];
    const float la = lsa[col],   lm = lsm[col];
#pragma unroll
    for (int j = 0; j < 8; ++j) {
      const int rl    = msub * 16 + j + ((lane < 16) ? 0 : 8);
      const float mu  = rowSum[rl] * invH;
      const float var = rowSq[rl] * invH - mu * mu;
      const float rs  = rsqrtf(var + 1e-12f);
      float v = (acc[t][j] - mu) * rs * g + be;
      v += pooled[rl * H + col];
      const float z = v - m1;
      v = ((z >= 0.f) ? z : al * z) + m2;
      v = v * la + lm;
      out[(size_t)(mrow + rl) * H + col] = v;
    }
  }
}

// ---------------------------------------------------------------------------
extern "C" void kernel_launch(void* const* d_in, const int* in_sizes, int n_in,
                              void* d_out, int out_size, void* d_ws, size_t ws_size,
                              hipStream_t stream) {
  const float* hs    = (const float*)d_in[0];   // [64,577,3072]
  const float* W     = (const float*)d_in[1];   // [768,3072]
  const float* b     = (const float*)d_in[2];   // [768]
  const float* move  = (const float*)d_in[3];   // [3072]
  const float* gamma = (const float*)d_in[4];
  const float* beta  = (const float*)d_in[5];
  const float* move1 = (const float*)d_in[6];
  const float* pa    = (const float*)d_in[7];
  const float* move2 = (const float*)d_in[8];
  const float* lsa   = (const float*)d_in[9];
  const float* lsm   = (const float*)d_in[10];
  float* out = (float*)d_out;

  unsigned short* Wp = (unsigned short*)d_ws;   // 768*3072*2 = 4.7 MB

  // 1) pack W -> bf16 fragments (L2-resident for the GEMM)
  {
    const int threads = 96 * 48 * 32;           // 147456
    pack_w_bf16<<<threads / 256, 256, 0, stream>>>(W, Wp);
  }

  // 2) fused GEMM + LN + pool + RPReLU + LayerScale
  {
    const size_t smem = (size_t)32 * 768 * 4      // pooled
                      + (size_t)2 * 32 * 40 * 2   // A double buffer (bf16)
                      + (size_t)64 * 4;           // row sums / sumsq
    const int nwg = 36928 / 32;                   // 1154 workgroups
    vit_out_fused<<<nwg, 256, smem, stream>>>(hs, Wp, b, move, gamma, beta,
                                              move1, pa, move2, lsa, lsm, out);
  }
}
